// PointTransformerSegment_1400159339079
// MI455X (gfx1250) — matrix operand order repaired and checked by reading, not compiled, not run on hardware
//
#include <hip/hip_runtime.h>

typedef __attribute__((ext_vector_type(16))) _Float16 v16h;
typedef __attribute__((ext_vector_type(8)))  float    v8f;

#define B_ 2
#define N_ 4096
#define D_ 128
#define H_ 12
#define K_ 16
#define LN_EPS 1e-5f

// ---- async global->LDS staging (gfx1250), guarded fallback -----------------
__device__ inline void stage_lds_b128(void* dst_lds, const void* src_glb,
                                      int nbytes, int tid, int nthreads) {
#if defined(__gfx1250__) && __has_builtin(__builtin_amdgcn_global_load_async_to_lds_b128)
  // builtin signature (from hipcc diagnostic): (int4 AS1*, int4 AS3*, Ii, Ii)
  typedef int v4i_gcc __attribute__((vector_size(16)));
  typedef __attribute__((address_space(1))) v4i_gcc* glb_p;
  typedef __attribute__((address_space(3))) v4i_gcc* lds_p;
  for (int off = tid * 16; off < nbytes; off += nthreads * 16) {
    __builtin_amdgcn_global_load_async_to_lds_b128(
        (glb_p)((unsigned long long)(uintptr_t)src_glb + (unsigned long long)off),
        (lds_p)(unsigned)((uintptr_t)dst_lds + (uintptr_t)off), 0, 0);
  }
#else
  for (int off = tid * 16; off < nbytes; off += nthreads * 16) {
    *(uint4*)((char*)dst_lds + off) = *(const uint4*)((const char*)src_glb + off);
  }
#endif
}

__device__ inline void wait_lds_stage() {
#if defined(__gfx1250__) && __has_builtin(__builtin_amdgcn_global_load_async_to_lds_b128)
#if __has_builtin(__builtin_amdgcn_s_wait_asynccnt)
  __builtin_amdgcn_s_wait_asynccnt(0);
#else
  asm volatile("s_wait_asynccnt 0" ::: "memory");
#endif
#endif
}

// ---- WMMA helpers (gfx1250 wave32, v_wmma_f32_16x16x32_f16) ----------------
// A fragment: 16x32 f16. lane = ph*16 + row, row = M. Per ISA layout:
//   lanes 0-15:  VGPR0-3 K=0..7,  VGPR4-7 K=16..23
//   lanes 16-31: VGPR0-3 K=8..15, VGPR4-7 K=24..31
__device__ inline v16h load_a_frag(const _Float16* A, int row, int ph, int k0) {
  v16h a;
#pragma unroll
  for (int e = 0; e < 16; ++e) {
    int j = e >> 1, pbit = e & 1;
    int k = (j < 4) ? (k0 + ph * 8 + 2 * j + pbit)
                    : (k0 + 16 + ph * 8 + 2 * (j - 4) + pbit);
    a[e] = A[row * D_ + k];
  }
  return a;
}

// B fragment: 32x16 f16, B[k][n] = W[n][k] (W row-major [out][in]).
//   lanes 0-15 hold K=k0+0..7,16..23, lanes 16-31 hold K=k0+8..15,24..31; N = lane&15.
__device__ inline v16h load_b_frag(const _Float16* W, int col, int ph, int k0, int nt) {
  v16h b;
  int n = nt * 16 + col;
  int kb = k0 + ph * 8;
#pragma unroll
  for (int e = 0; e < 16; ++e) b[e] = W[n * D_ + kb + (e < 8 ? e : e + 8)];
  return b;
}

__device__ inline v8f wmma32(v16h a, v16h b, v8f c) {
  return __builtin_amdgcn_wmma_f32_16x16x32_f16(false, a, false, b, (short)0, c,
                                                false, false);
}

// ---- Kernel 0: one-shot f32 -> f16 weight conversion -----------------------
__global__ __launch_bounds__(256) void cvt_weights_kernel(
    const float* __restrict__ wq, const float* __restrict__ wk,
    const float* __restrict__ wv, const float* __restrict__ aw1,
    const float* __restrict__ aw2, _Float16* __restrict__ wqh,
    _Float16* __restrict__ wkh, _Float16* __restrict__ wvh,
    _Float16* __restrict__ aw1h, _Float16* __restrict__ aw2h) {
  int i = blockIdx.x * 256 + threadIdx.x;
  int m = blockIdx.y;
  const float* s = (m == 0) ? wq : (m == 1) ? wk : (m == 2) ? wv
                  : (m == 3) ? aw1 : aw2;
  _Float16* d = (m == 0) ? wqh : (m == 1) ? wkh : (m == 2) ? wvh
               : (m == 3) ? aw1h : aw2h;
  d[i] = (_Float16)s[i];
}

// ---- Kernel 1: KNN (top-16 by squared distance, ascending, stable ties) ----
__global__ __launch_bounds__(128) void knn_kernel(const float* __restrict__ pos,
                                                  int* __restrict__ idx_out) {
  int b = blockIdx.y;
  int q = blockIdx.x * 128 + threadIdx.x;
  const float* pb = pos + (size_t)b * N_ * 3;
  float qx = pb[q * 3 + 0], qy = pb[q * 3 + 1], qz = pb[q * 3 + 2];
  float bd[K_];
  int bi[K_];
#pragma unroll
  for (int s = 0; s < K_; ++s) { bd[s] = 3.0e38f; bi[s] = 0; }
  __shared__ float cp[128 * 3];
  for (int t = 0; t < N_ / 128; ++t) {
    int c0 = t * 128;
    cp[threadIdx.x * 3 + 0] = pb[(c0 + threadIdx.x) * 3 + 0];
    cp[threadIdx.x * 3 + 1] = pb[(c0 + threadIdx.x) * 3 + 1];
    cp[threadIdx.x * 3 + 2] = pb[(c0 + threadIdx.x) * 3 + 2];
    __syncthreads();
    for (int j = 0; j < 128; ++j) {
      float dx = qx - cp[j * 3 + 0];
      float dy = qy - cp[j * 3 + 1];
      float dz = qz - cp[j * 3 + 2];
      float d2 = dx * dx + dy * dy + dz * dz;
      if (d2 < bd[K_ - 1]) {
        bd[K_ - 1] = d2;
        bi[K_ - 1] = c0 + j;
#pragma unroll
        for (int s = K_ - 1; s > 0; --s) {
          if (bd[s] < bd[s - 1]) {
            float td = bd[s]; bd[s] = bd[s - 1]; bd[s - 1] = td;
            int ti = bi[s]; bi[s] = bi[s - 1]; bi[s - 1] = ti;
          }
        }
      }
    }
    __syncthreads();
  }
  int* o = idx_out + ((size_t)b * N_ + q) * K_;
#pragma unroll
  for (int s = 0; s < K_; ++s) o[s] = bi[s];
}

// ---- Kernel 2: q/k/v projections, x f32 in -> f16 tables in workspace ------
__global__ __launch_bounds__(128) void qkv_kernel(
    const float* __restrict__ x, const _Float16* __restrict__ wqh,
    const _Float16* __restrict__ wkh, const _Float16* __restrict__ wvh,
    _Float16* __restrict__ Q, _Float16* __restrict__ Km,
    _Float16* __restrict__ Vm) {
  __shared__ __align__(16) _Float16 wbuf[D_ * D_];
  __shared__ __align__(16) _Float16 As[4][16 * D_];
  int tid = threadIdx.x, lane = tid & 31, wid = tid >> 5;
  int m = blockIdx.y;
  const _Float16* W = (m == 0) ? wqh : ((m == 1) ? wkh : wvh);
  _Float16* O = (m == 0) ? Q : ((m == 1) ? Km : Vm);
  stage_lds_b128(wbuf, W, D_ * D_ * 2, tid, 128);
  int rowbase = blockIdx.x * 64 + wid * 16;
  for (int i = lane; i < 16 * D_; i += 32) {
    int r = i >> 7, c = i & 127;
    As[wid][i] = (_Float16)x[(size_t)(rowbase + r) * D_ + c];
  }
  wait_lds_stage();
  __syncthreads();
  int col0 = lane & 15, ph = lane >> 4;
  v16h af[4];
#pragma unroll
  for (int kk = 0; kk < 4; ++kk) af[kk] = load_a_frag(As[wid], col0, ph, kk * 32);
#pragma unroll
  for (int nt = 0; nt < 8; ++nt) {
    v8f acc;
#pragma unroll
    for (int r = 0; r < 8; ++r) acc[r] = 0.f;
#pragma unroll
    for (int kk = 0; kk < 4; ++kk) {
      v16h bf = load_b_frag(wbuf, col0, ph, kk * 32, nt);
      acc = wmma32(af[kk], bf, acc);
    }
#pragma unroll
    for (int r = 0; r < 8; ++r) {
      int row = rowbase + r + 8 * ph;
      int col = nt * 16 + col0;
      O[(size_t)row * D_ + col] = (_Float16)acc[r];
    }
  }
}

// ---- Kernel 3: fused gather + pos_mlp + attn_mlp + softmax + aggregate -----
// 128 threads = 4 waves; each wave owns one point (16 neighbors = 16 WMMA rows).
__global__ __launch_bounds__(128) void fused_attn_kernel(
    const float* __restrict__ pos, const float* __restrict__ pw1,
    const float* __restrict__ pb1, const float* __restrict__ pg,
    const float* __restrict__ pbeta, const float* __restrict__ pw2,
    const float* __restrict__ pb2, const _Float16* __restrict__ aw1h,
    const float* __restrict__ ab1, const float* __restrict__ ag,
    const float* __restrict__ abeta, const _Float16* __restrict__ aw2h,
    const float* __restrict__ ab2, const _Float16* __restrict__ Q,
    const _Float16* __restrict__ Km, const _Float16* __restrict__ Vm,
    const int* __restrict__ knn, float* __restrict__ out) {
  __shared__ __align__(16) _Float16 wbuf[D_ * D_];    // w1 then w2 (reused)
  __shared__ __align__(16) _Float16 Abuf[4][16 * D_]; // per-wave A tile
  __shared__ float pw2s[D_ * H_];
  __shared__ float hbuf[4][16][H_];
  __shared__ int idxb[4][16];
  __shared__ float b1s[D_], gs[D_], bes[D_], b2s[D_], pb2s[D_];

  int tid = threadIdx.x, lane = tid & 31, wid = tid >> 5;
  int col0 = lane & 15, ph = lane >> 4;
  int p = blockIdx.x * 4 + wid;
  int b = p >> 12, n = p & (N_ - 1);

  // stage w1 (f16, async path) + small tables; prefetch w2 toward L2/WGP$
  stage_lds_b128(wbuf, aw1h, D_ * D_ * 2, tid, 128);
  __builtin_prefetch((const char*)aw2h + tid * 256, 0, 1);
  __builtin_prefetch((const char*)aw2h + tid * 256 + 128, 0, 1);
  for (int i = tid; i < D_ * H_; i += 128) pw2s[i] = pw2[i];
  b1s[tid] = ab1[tid]; gs[tid] = ag[tid]; bes[tid] = abeta[tid];
  b2s[tid] = ab2[tid]; pb2s[tid] = pb2[tid];
  if (lane < K_) idxb[wid][lane] = knn[(size_t)p * K_ + lane];
  wait_lds_stage();
  __syncthreads();

  // (a) pos_mlp hidden: Linear(3,12)+LN+ReLU, one neighbor per lane 0..15
  if (lane < K_) {
    int g = idxb[wid][lane];
    const float* p0 = pos + ((size_t)b * N_ + n) * 3;
    const float* p1 = pos + ((size_t)b * N_ + g) * 3;
    float rx = p0[0] - p1[0], ry = p0[1] - p1[1], rz = p0[2] - p1[2];
    float h[H_];
    float s = 0.f;
#pragma unroll
    for (int m = 0; m < H_; ++m) {
      float v = rx * pw1[m * 3 + 0] + ry * pw1[m * 3 + 1] + rz * pw1[m * 3 + 2] +
                pb1[m];
      h[m] = v; s += v;
    }
    float mu = s * (1.f / H_);
    float sq = 0.f;
#pragma unroll
    for (int m = 0; m < H_; ++m) { float d = h[m] - mu; sq += d * d; }
    float inv = rsqrtf(sq * (1.f / H_) + LN_EPS);
#pragma unroll
    for (int m = 0; m < H_; ++m) {
      float v = (h[m] - mu) * inv * pg[m] + pbeta[m];
      hbuf[wid][lane][m] = fmaxf(v, 0.f);
    }
  }
  __syncthreads();

  // (b) pe = Linear(12,128); A = q - kf + pe (f16, LDS); S = vf + pe (regs,
  // held exactly at the D-matrix positions needed for the softmax stage)
  float sreg[64];
#pragma unroll
  for (int nt = 0; nt < 8; ++nt) {
    int d = nt * 16 + col0;
    float pw[H_];
#pragma unroll
    for (int m = 0; m < H_; ++m) pw[m] = pw2s[d * H_ + m];
    float pb2v = pb2s[d];
    float qd = (float)Q[(size_t)p * D_ + d];
#pragma unroll
    for (int r = 0; r < 8; ++r) {
      int k = r + 8 * ph;
      int g = idxb[wid][k];
      size_t grow = (size_t)b * N_ + g;
      float pe = pb2v;
#pragma unroll
      for (int m = 0; m < H_; ++m) pe += hbuf[wid][k][m] * pw[m];
      float kf = (float)Km[grow * D_ + d];
      float vf = (float)Vm[grow * D_ + d];
      Abuf[wid][k * D_ + d] = (_Float16)(qd - kf + pe);
      sreg[nt * 8 + r] = vf + pe;
    }
  }
  __syncthreads();

  // (c) GEMM1 (16x128 x 128x128) + bias + LayerNorm (shuffle stats) + ReLU
  v16h af[4];
#pragma unroll
  for (int kk = 0; kk < 4; ++kk) af[kk] = load_a_frag(Abuf[wid], col0, ph, kk * 32);
  v8f acc[8];
#pragma unroll
  for (int nt = 0; nt < 8; ++nt) {
#pragma unroll
    for (int r = 0; r < 8; ++r) acc[nt][r] = 0.f;
#pragma unroll
    for (int kk = 0; kk < 4; ++kk) {
      v16h bf = load_b_frag(wbuf, col0, ph, kk * 32, nt);
      acc[nt] = wmma32(af[kk], bf, acc[nt]);
    }
  }
  float rs[8], rq[8];
#pragma unroll
  for (int r = 0; r < 8; ++r) { rs[r] = 0.f; rq[r] = 0.f; }
#pragma unroll
  for (int nt = 0; nt < 8; ++nt) {
#pragma unroll
    for (int r = 0; r < 8; ++r) {
      float v = acc[nt][r] + b1s[nt * 16 + col0];
      acc[nt][r] = v; rs[r] += v; rq[r] += v * v;
    }
  }
#pragma unroll
  for (int r = 0; r < 8; ++r) {
#pragma unroll
    for (int m = 1; m < 16; m <<= 1) {  // stays within each 16-lane half
      rs[r] += __shfl_xor(rs[r], m, 32);
      rq[r] += __shfl_xor(rq[r], m, 32);
    }
  }
  float mu[8], inv[8];
#pragma unroll
  for (int r = 0; r < 8; ++r) {
    mu[r] = rs[r] * (1.f / D_);
    float var = rq[r] * (1.f / D_) - mu[r] * mu[r];
    inv[r] = rsqrtf(var + LN_EPS);
  }
#pragma unroll
  for (int nt = 0; nt < 8; ++nt) {
#pragma unroll
    for (int r = 0; r < 8; ++r) {
      int col = nt * 16 + col0;
      float v = (acc[nt][r] - mu[r]) * inv[r] * gs[col] + bes[col];
      Abuf[wid][(r + 8 * ph) * D_ + col] = (_Float16)fmaxf(v, 0.f);
    }
  }
  __syncthreads();

  // stage w2 (f16, async path) into the same LDS weight buffer
  stage_lds_b128(wbuf, aw2h, D_ * D_ * 2, tid, 128);
  wait_lds_stage();
  __syncthreads();

  // (d) GEMM2
#pragma unroll
  for (int kk = 0; kk < 4; ++kk) af[kk] = load_a_frag(Abuf[wid], col0, ph, kk * 32);
#pragma unroll
  for (int nt = 0; nt < 8; ++nt) {
#pragma unroll
    for (int r = 0; r < 8; ++r) acc[nt][r] = 0.f;
#pragma unroll
    for (int kk = 0; kk < 4; ++kk) {
      v16h bf = load_b_frag(wbuf, col0, ph, kk * 32, nt);
      acc[nt] = wmma32(af[kk], bf, acc[nt]);
    }
  }

  // (e) softmax over the 16 neighbors (rows) per channel + weighted sum of S
#pragma unroll
  for (int nt = 0; nt < 8; ++nt) {
    int d = nt * 16 + col0;
    float b2v = b2s[d];
    float l[8];
    float mx = -3.0e38f;
#pragma unroll
    for (int r = 0; r < 8; ++r) { l[r] = acc[nt][r] + b2v; mx = fmaxf(mx, l[r]); }
    mx = fmaxf(mx, __shfl_xor(mx, 16, 32));  // combine the two 8-row halves
    float es = 0.f, ps = 0.f;
#pragma unroll
    for (int r = 0; r < 8; ++r) {
      float e = __expf(l[r] - mx);
      es += e;
      ps += e * sreg[nt * 8 + r];
    }
    es += __shfl_xor(es, 16, 32);
    ps += __shfl_xor(ps, 16, 32);
    if (ph == 0) out[(size_t)p * D_ + d] = ps / es;
  }
}

extern "C" void kernel_launch(void* const* d_in, const int* in_sizes, int n_in,
                              void* d_out, int out_size, void* d_ws,
                              size_t ws_size, hipStream_t stream) {
  (void)in_sizes; (void)n_in; (void)out_size; (void)ws_size;
  const float* x     = (const float*)d_in[0];
  const float* pos   = (const float*)d_in[1];
  const float* w_q   = (const float*)d_in[2];
  const float* w_k   = (const float*)d_in[3];
  const float* w_v   = (const float*)d_in[4];
  const float* pw1   = (const float*)d_in[5];
  const float* pb1   = (const float*)d_in[6];
  const float* pg    = (const float*)d_in[7];
  const float* pbeta = (const float*)d_in[8];
  const float* pw2   = (const float*)d_in[9];
  const float* pb2   = (const float*)d_in[10];
  const float* aw1   = (const float*)d_in[11];
  const float* ab1   = (const float*)d_in[12];
  const float* ag    = (const float*)d_in[13];
  const float* abeta = (const float*)d_in[14];
  const float* aw2   = (const float*)d_in[15];
  const float* ab2   = (const float*)d_in[16];
  float* out = (float*)d_out;

  // workspace: Q | K | V (f16) | knn idx | 5x f16 weight copies  (~6.8 MB)
  _Float16* Q  = (_Float16*)d_ws;
  _Float16* Km = Q  + (size_t)B_ * N_ * D_;
  _Float16* Vm = Km + (size_t)B_ * N_ * D_;
  int* knn_idx = (int*)(Vm + (size_t)B_ * N_ * D_);
  _Float16* wqh  = (_Float16*)(knn_idx + (size_t)B_ * N_ * K_);
  _Float16* wkh  = wqh  + D_ * D_;
  _Float16* wvh  = wkh  + D_ * D_;
  _Float16* aw1h = wvh  + D_ * D_;
  _Float16* aw2h = aw1h + D_ * D_;

  cvt_weights_kernel<<<dim3((D_ * D_) / 256, 5), 256, 0, stream>>>(
      w_q, w_k, w_v, aw1, aw2, wqh, wkh, wvh, aw1h, aw2h);
  knn_kernel<<<dim3(N_ / 128, B_), 128, 0, stream>>>(pos, knn_idx);
  qkv_kernel<<<dim3((B_ * N_) / 64, 3), 128, 0, stream>>>(x, wqh, wkh, wvh, Q,
                                                          Km, Vm);
  fused_attn_kernel<<<dim3((B_ * N_) / 4), 128, 0, stream>>>(
      pos, pw1, pb1, pg, pbeta, pw2, pb2, aw1h, ab1, ag, abeta, aw2h, ab2, Q,
      Km, Vm, knn_idx, out);
}
